// DiT_63866163691579
// MI455X (gfx1250) — matrix operand order repaired and checked by reading, compiled, not run on hardware
//
#include <hip/hip_runtime.h>
#include <hip/hip_bf16.h>

// ---------------------------------------------------------------------------
// Types for CDNA5 WMMA (wave32)
// ---------------------------------------------------------------------------
typedef __bf16 bf16_t;
typedef bf16_t bf16x8  __attribute__((ext_vector_type(8)));
typedef bf16_t bf16x16 __attribute__((ext_vector_type(16)));
typedef float  f32x8   __attribute__((ext_vector_type(8)));

static __device__ __forceinline__ f32x8 wmma_bf16(bf16x16 a, bf16x16 b, f32x8 c) {
    // D = A(16x32 bf16) * B(32x16 bf16) + C(16x16 f32)
    return __builtin_amdgcn_wmma_f32_16x16x32_bf16(
        /*neg_a=*/false, a, /*neg_b=*/false, b,
        /*c_mod=*/(short)0, c, /*reuse_a=*/false, /*reuse_b=*/false);
}

static __device__ __forceinline__ bf16_t f2bf(float f) {
    unsigned u = __builtin_bit_cast(unsigned, f);
    unsigned r = u + 0x7fffu + ((u >> 16) & 1u);          // round-to-nearest-even
    unsigned short s = (unsigned short)(r >> 16);
    return __builtin_bit_cast(bf16_t, s);
}

// CDNA5 async copy: global -> LDS, tracked by ASYNCcnt (no VGPR writeback).
// lds_off = wave-relative LDS byte offset (low 32 bits of flat shared address).
static __device__ __forceinline__ void async_copy_b128(unsigned lds_off,
                                                       unsigned long long gaddr) {
    asm volatile("global_load_async_to_lds_b128 %0, %1, off"
                 :: "v"(lds_off), "v"(gaddr) : "memory");
}
static __device__ __forceinline__ void wait_asynccnt0() {
#if __has_builtin(__builtin_amdgcn_s_wait_asynccnt)
    __builtin_amdgcn_s_wait_asynccnt(0);
#else
    asm volatile("s_wait_asynccnt 0x0" ::: "memory");
#endif
}

// A-operand fragment: 16(M) x 32(K) bf16.
// lane L: row = L&15 ; K chunks [h8, h8+8) and [16+h8, 16+h8+8), h8 = (L>>4)*8
static __device__ __forceinline__ bf16x16 load_frag_a(const bf16_t* __restrict__ base,
                                                      int ld, int lane) {
    const int r  = lane & 15;
    const int h8 = (lane >> 4) * 8;
    const bf16_t* p = base + (size_t)r * ld;
    bf16x8 lo = *(const bf16x8*)(p + h8);
    bf16x8 hi = *(const bf16x8*)(p + 16 + h8);
    return __builtin_shufflevector(lo, hi, 0,1,2,3,4,5,6,7,8,9,10,11,12,13,14,15);
}

// B-operand fragment: 32(K) x 16(N) bf16, stored K-major per column (row-major NxK).
// lane L: col = L&15 ; K chunks [h8, h8+8) and [16+h8, 16+h8+8), h8 = (L>>4)*8
static __device__ __forceinline__ bf16x16 load_frag_b(const bf16_t* __restrict__ base,
                                                      int ld, int lane) {
    const int n   = lane & 15;
    const int h16 = (lane >> 4) * 8;
    const bf16_t* p = base + (size_t)n * ld + h16;
    bf16x8 lo = *(const bf16x8*)(p);
    bf16x8 hi = *(const bf16x8*)(p + 16);
    return __builtin_shufflevector(lo, hi, 0,1,2,3,4,5,6,7,8,9,10,11,12,13,14,15);
}

static __device__ __forceinline__ float gelu_exact(float v) {
    return 0.5f * v * (1.0f + erff(v * 0.70710678118654752f));
}
static __device__ __forceinline__ float sigmoidf(float v) {
    return 1.0f / (1.0f + __expf(-v));
}

// ---------------------------------------------------------------------------
// Elementwise f32 -> bf16 conversion
// ---------------------------------------------------------------------------
__global__ __launch_bounds__(256) void cvt_bf16_kernel(const float* __restrict__ in,
                                                       bf16_t* __restrict__ out, int n) {
    int i = blockIdx.x * 256 + threadIdx.x;
    if (i < n) out[i] = f2bf(in[i]);
}

// ---------------------------------------------------------------------------
// LayerNorm over D=1024, bf16 output. One row per block (256 threads, 4 el/thread).
// ---------------------------------------------------------------------------
__global__ __launch_bounds__(256) void ln_bf16_kernel(const float* __restrict__ x,
                                                      const float* __restrict__ g,
                                                      const float* __restrict__ b,
                                                      bf16_t* __restrict__ out) {
    const int row = blockIdx.x;
    const int t   = threadIdx.x;
    const float* xr = x + (size_t)row * 1024;
    float4 v = ((const float4*)xr)[t];
    float s  = v.x + v.y + v.z + v.w;
    float ss = v.x * v.x + v.y * v.y + v.z * v.z + v.w * v.w;

    __shared__ float sh1[256], sh2[256];
    sh1[t] = s; sh2[t] = ss;
    __syncthreads();
    for (int st = 128; st > 0; st >>= 1) {
        if (t < st) { sh1[t] += sh1[t + st]; sh2[t] += sh2[t + st]; }
        __syncthreads();
    }
    const float mean = sh1[0] * (1.0f / 1024.0f);
    const float var  = sh2[0] * (1.0f / 1024.0f) - mean * mean;
    const float rstd = rsqrtf(var + 1e-5f);

    bf16_t* o = out + (size_t)row * 1024;
    const int c = t * 4;
    o[c + 0] = f2bf((v.x - mean) * rstd * g[c + 0] + b[c + 0]);
    o[c + 1] = f2bf((v.y - mean) * rstd * g[c + 1] + b[c + 1]);
    o[c + 2] = f2bf((v.z - mean) * rstd * g[c + 2] + b[c + 2]);
    o[c + 3] = f2bf((v.w - mean) * rstd * g[c + 3] + b[c + 3]);
}

// ---------------------------------------------------------------------------
// WMMA GEMM: C[M,N] = A[M,K](bf16) @ W[N,K]^T(bf16) + bias, fused epilogues.
// Block: 256 threads (8 waves) -> 128x128 C tile. Wave: 64x32 (4x2 WMMA tiles).
// A tile (128x32) staged in LDS via double-buffered GLOBAL_LOAD_ASYNC_TO_LDS_B128
// (ASYNCcnt overlap); W fragments streamed straight from L2 (weights resident).
// LDS rows padded to 40 elems (20-bank stride -> conflict-free ds_read_b128).
// EPI: 0 = f32 store            1 = bf16 store
//      2 = GELU -> f32          3 = sigmoid * aux -> bf16 (gate fusion)
//      4 = + aux (residual) -> f32
// ---------------------------------------------------------------------------
#define LDSA_LD 40

template <int EPI>
__global__ __launch_bounds__(256)
void gemm_bf16_kernel(const bf16_t* __restrict__ A, const bf16_t* __restrict__ W,
                      const float* __restrict__ bias, const float* __restrict__ aux,
                      float* __restrict__ outF, bf16_t* __restrict__ outB,
                      int M, int N, int K) {
    __shared__ bf16_t ldsA[2][128][LDSA_LD];

    const int tid  = threadIdx.x;
    const int lane = tid & 31;
    const int wave = tid >> 5;
    const int wm   = wave & 1;
    const int wn   = wave >> 1;
    const int mblk = blockIdx.y * 128;
    const int m0   = mblk + wm * 64;
    const int n0   = blockIdx.x * 128 + wn * 32;

    // Stage one 128x32 A tile into LDS buffer `buf` (2 async b128 per thread).
    auto stage = [&](int buf, int kt) {
#pragma unroll
        for (int s = 0; s < 2; ++s) {
            const int slot = tid + s * 256;          // 512 x 16B chunks = 8 KB
            const int row  = slot >> 2;
            const int ch   = (slot & 3) * 8;         // bf16 element offset in row
            const bf16_t* gp = A + (size_t)(mblk + row) * K + kt + ch;
            const unsigned la = (unsigned)(size_t)&ldsA[buf][row][ch];
            async_copy_b128(la, (unsigned long long)(size_t)gp);
        }
    };

    f32x8 acc[4][2];
    for (int mt = 0; mt < 4; ++mt)
        for (int nt = 0; nt < 2; ++nt)
            for (int e = 0; e < 8; ++e) acc[mt][nt][e] = 0.0f;

    stage(0, 0);

    for (int k = 0; k < K; k += 32) {
        const int cur = (k >> 5) & 1;
        wait_asynccnt0();          // this wave's fill of buf[cur] complete
        __syncthreads();           // all waves' fills visible
        if (k + 32 < K) stage(cur ^ 1, k + 32);
        if (k + 64 < K)
            __builtin_prefetch(W + (size_t)n0 * K + k + 64, 0, 1);

        bf16x16 af[4];
#pragma unroll
        for (int mt = 0; mt < 4; ++mt)
            af[mt] = load_frag_a(&ldsA[cur][wm * 64 + mt * 16][0], LDSA_LD, lane);
        bf16x16 wf[2];
#pragma unroll
        for (int nt = 0; nt < 2; ++nt)
            wf[nt] = load_frag_b(W + (size_t)(n0 + nt * 16) * K + k, K, lane);
#pragma unroll
        for (int mt = 0; mt < 4; ++mt)
#pragma unroll
            for (int nt = 0; nt < 2; ++nt)
                acc[mt][nt] = wmma_bf16(af[mt], wf[nt], acc[mt][nt]);
    }

    const int nl  = lane & 15;
    const int hlf = lane >> 4;
#pragma unroll
    for (int mt = 0; mt < 4; ++mt)
#pragma unroll
        for (int nt = 0; nt < 2; ++nt)
#pragma unroll
            for (int i = 0; i < 8; ++i) {
                const int row = m0 + mt * 16 + hlf * 8 + i;
                const int col = n0 + nt * 16 + nl;
                float v = acc[mt][nt][i] + bias[col];
                const size_t o = (size_t)row * N + col;
                if      (EPI == 0) outF[o] = v;
                else if (EPI == 1) outB[o] = f2bf(v);
                else if (EPI == 2) outF[o] = gelu_exact(v);
                else if (EPI == 3) outB[o] = f2bf(sigmoidf(v) * aux[o]);
                else if (EPI == 4) outF[o] = v + aux[o];
            }
}

// ---------------------------------------------------------------------------
// RoPE + head layout prep for self-attention.
// qkv: [B*S, 3072] f32. Outputs: qb,kb [B,S,H*DH] bf16 (rope'd), vT [B,H,DH,S] bf16.
// ---------------------------------------------------------------------------
__global__ __launch_bounds__(256)
void rope_prep_kernel(const float* __restrict__ qkv, const float* __restrict__ rot,
                      bf16_t* __restrict__ qb, bf16_t* __restrict__ kb,
                      bf16_t* __restrict__ vT) {
    const int idx = blockIdx.x * 256 + threadIdx.x;      // over B*S*D = 4M
    const int d  = idx & 1023;
    const int s  = (idx >> 10) & 1023;
    const int bi = idx >> 20;
    const int h  = d >> 6;
    const int dh = d & 63;

    const float* r = qkv + (size_t)(bi * 1024 + s) * 3072;
    const float qv = r[d];
    const float kv = r[1024 + d];
    const float vv = r[2048 + d];

    const float pos = rot[s * 64 + dh];
    const float cs = __cosf(pos), sn = __sinf(pos);
    const int   po = (dh < 32) ? (d + 32) : (d - 32);
    const float qrh = (dh < 32) ? -r[po]        : r[po];
    const float krh = (dh < 32) ? -r[1024 + po] : r[1024 + po];

    qb[idx] = f2bf(qv * cs + qrh * sn);
    kb[idx] = f2bf(kv * cs + krh * sn);
    vT[(((size_t)bi * 16 + h) * 64 + dh) * 1024 + s] = f2bf(vv);
}

// Cross-attention V transpose: kvb [B*256, 2048] bf16 (k|v) -> vT [B,H,DH,256] bf16
__global__ __launch_bounds__(256)
void cross_vt_kernel(const bf16_t* __restrict__ kvb, bf16_t* __restrict__ vT) {
    const int idx = blockIdx.x * 256 + threadIdx.x;      // over B*TLEN*D = 1M
    const int d  = idx & 1023;
    const int t  = (idx >> 10) & 255;
    const int bi = idx >> 18;
    const int h  = d >> 6;
    const int dh = d & 63;
    vT[(((size_t)bi * 16 + h) * 64 + dh) * 256 + t] =
        kvb[(size_t)(bi * 256 + t) * 2048 + 1024 + d];
}

// ---------------------------------------------------------------------------
// Flash-style attention, WMMA everywhere. 8 waves/block, 16 query rows per wave.
// Q: [B, Sq, 16*64] bf16 (ld=1024). K: rows are keys, ld=ldk, head offset h*64.
// vT: [B,H,DH,Skv] bf16. kscale = (1/sqrt(DH)) * log2(e).
// EPI: 0 -> f32 out + residual ; 1 -> bf16 out
// ---------------------------------------------------------------------------
template <int EPI>
__global__ __launch_bounds__(256)
void attn_kernel(const bf16_t* __restrict__ Qb, const bf16_t* __restrict__ Kb,
                 const bf16_t* __restrict__ vT, const float* __restrict__ resid,
                 float* __restrict__ outF, bf16_t* __restrict__ outB,
                 int Sq, int Skv, int ldk, long kb_batch_stride, float kscale) {
    __shared__ bf16_t lds_p[8][16][32];                  // per-wave P tile staging

    const int tid  = threadIdx.x;
    const int lane = tid & 31;
    const int wave = tid >> 5;
    const int bh   = blockIdx.y;
    const int bi   = bh >> 4;
    const int h    = bh & 15;
    const int s0   = blockIdx.x * 128 + wave * 16;

    const bf16_t* qbase = Qb + (size_t)bi * Sq * 1024 + (size_t)h * 64;
    const bf16_t* kbase = Kb + (size_t)bi * kb_batch_stride + (size_t)h * 64;
    const bf16_t* vbase = vT + ((size_t)(bi * 16 + h) * 64) * Skv;

    const bf16x16 aq0 = load_frag_a(qbase + (size_t)s0 * 1024,      1024, lane);
    const bf16x16 aq1 = load_frag_a(qbase + (size_t)s0 * 1024 + 32, 1024, lane);

    f32x8 accO[4];
    for (int n = 0; n < 4; ++n)
        for (int e = 0; e < 8; ++e) accO[n][e] = 0.0f;
    float mrow[8], lrow[8];
#pragma unroll
    for (int i = 0; i < 8; ++i) { mrow[i] = -1e30f; lrow[i] = 0.0f; }

    const int nl  = lane & 15;
    const int hlf = lane >> 4;

    for (int kt = 0; kt < Skv; kt += 32) {
        // S tile (16 q rows x 32 keys) via 4 WMMAs
        f32x8 sc[2];
#pragma unroll
        for (int j = 0; j < 2; ++j) {
            bf16x16 bk0 = load_frag_b(kbase + (size_t)(kt + j * 16) * ldk,      ldk, lane);
            bf16x16 bk1 = load_frag_b(kbase + (size_t)(kt + j * 16) * ldk + 32, ldk, lane);
            f32x8 c;
            for (int e = 0; e < 8; ++e) c[e] = 0.0f;
            c = wmma_bf16(aq0, bk0, c);
            c = wmma_bf16(aq1, bk1, c);
            sc[j] = c;
        }
        // online softmax (rows live in 16-lane halves of the wave)
#pragma unroll
        for (int i = 0; i < 8; ++i) {
            float t0 = sc[0][i] * kscale;
            float t1 = sc[1][i] * kscale;
            float mx = fmaxf(t0, t1);
            for (int off = 8; off > 0; off >>= 1) mx = fmaxf(mx, __shfl_xor(mx, off));
            const float mnew = fmaxf(mrow[i], mx);
            const float corr = exp2f(mrow[i] - mnew);
            const float p0 = exp2f(t0 - mnew);
            const float p1 = exp2f(t1 - mnew);
            float ps = p0 + p1;
            for (int off = 8; off > 0; off >>= 1) ps += __shfl_xor(ps, off);
            lrow[i] = lrow[i] * corr + ps;
            mrow[i] = mnew;
#pragma unroll
            for (int n = 0; n < 4; ++n) accO[n][i] *= corr;
            const int row = hlf * 8 + i;
            lds_p[wave][row][nl]      = f2bf(p0);
            lds_p[wave][row][16 + nl] = f2bf(p1);
        }
        // P (16x32) @ V (32x64) via 4 WMMAs; P re-read from LDS in A layout
        const bf16x16 pf = load_frag_a(&lds_p[wave][0][0], 32, lane);
#pragma unroll
        for (int n = 0; n < 4; ++n) {
            bf16x16 bv = load_frag_b(vbase + (size_t)(n * 16) * Skv + kt, Skv, lane);
            accO[n] = wmma_bf16(pf, bv, accO[n]);
        }
    }

#pragma unroll
    for (int n = 0; n < 4; ++n)
#pragma unroll
        for (int i = 0; i < 8; ++i) {
            const int row = s0 + hlf * 8 + i;
            const int col = h * 64 + n * 16 + nl;
            const float v = accO[n][i] / lrow[i];
            const size_t o = ((size_t)bi * Sq + row) * 1024 + col;
            if (EPI == 0) outF[o] = v + resid[o];
            else          outB[o] = f2bf(v);
        }
}

// ---------------------------------------------------------------------------
// Host-side orchestration
// ---------------------------------------------------------------------------
static constexpr size_t MB = 1024ull * 1024ull;
// persistent
static constexpr size_t OFF_W_ATTN  = 0;                 // 6 MB  (3M bf16)
static constexpr size_t OFF_W_CAIN  = OFF_W_ATTN + 6 * MB;
static constexpr size_t OFF_W_CAOUT = OFF_W_CAIN + 6 * MB;   // 2 MB
static constexpr size_t OFF_W_FC1   = OFF_W_CAOUT + 2 * MB;  // 8 MB
static constexpr size_t OFF_W_GATE  = OFF_W_FC1 + 8 * MB;    // 8 MB
static constexpr size_t OFF_W_FC2   = OFF_W_GATE + 8 * MB;   // 8 MB
static constexpr size_t OFF_X1      = OFF_W_FC2 + 8 * MB;    // 16 MB f32
static constexpr size_t OFF_X2      = OFF_X1 + 16 * MB;      // 16 MB f32
static constexpr size_t OFF_XN      = OFF_X2 + 16 * MB;      // 8 MB bf16
static constexpr size_t OFF_TRANS   = OFF_XN + 8 * MB;
// phase A (self-attn)
static constexpr size_t OFF_QKV = OFF_TRANS;                 // 48 MB f32
static constexpr size_t OFF_QH  = OFF_QKV + 48 * MB;         // 8 MB bf16
static constexpr size_t OFF_KH  = OFF_QH + 8 * MB;           // 8 MB bf16
static constexpr size_t OFF_VT  = OFF_KH + 8 * MB;           // 8 MB bf16
// phase B (cross-attn, reuses A slots)
static constexpr size_t OFF_TBUF = OFF_TRANS;                // 2 MB bf16 (text)
static constexpr size_t OFF_OBUF = OFF_TRANS + 8 * MB;       // 8 MB bf16 (cross O)
static constexpr size_t OFF_QCB  = OFF_QH;                   // bf16 q (cross)
static constexpr size_t OFF_KVB  = OFF_KH;                   // 4 MB bf16 (k|v cross)
static constexpr size_t OFF_VTC  = OFF_VT;                   // 2 MB bf16
// phase C (MLP)
static constexpr size_t OFF_HBUF = OFF_TRANS;                // 64 MB f32
static constexpr size_t OFF_HGB  = OFF_TRANS + 64 * MB;      // 32 MB bf16

extern "C" void kernel_launch(void* const* d_in, const int* in_sizes, int n_in,
                              void* d_out, int out_size, void* d_ws, size_t ws_size,
                              hipStream_t stream) {
    (void)in_sizes; (void)n_in; (void)out_size; (void)ws_size;

    const float* x        = (const float*)d_in[0];
    const float* text_emb = (const float*)d_in[1];
    const float* rot      = (const float*)d_in[3];
    const float* ln1_g = (const float*)d_in[4],  *ln1_b = (const float*)d_in[5];
    const float* ln2_g = (const float*)d_in[6],  *ln2_b = (const float*)d_in[7];
    const float* ln3_g = (const float*)d_in[8],  *ln3_b = (const float*)d_in[9];
    const float* attn_in_w = (const float*)d_in[10], *attn_in_b = (const float*)d_in[11];
    const float* ca_in_w   = (const float*)d_in[12], *ca_in_b   = (const float*)d_in[13];
    const float* ca_out_w  = (const float*)d_in[14], *ca_out_b  = (const float*)d_in[15];
    const float* fc1_w  = (const float*)d_in[16], *fc1_b  = (const float*)d_in[17];
    const float* gate_w = (const float*)d_in[18], *gate_b = (const float*)d_in[19];
    const float* fc2_w  = (const float*)d_in[20], *fc2_b  = (const float*)d_in[21];
    float* out = (float*)d_out;

    char* ws = (char*)d_ws;
    bf16_t* wAttn  = (bf16_t*)(ws + OFF_W_ATTN);
    bf16_t* wCaIn  = (bf16_t*)(ws + OFF_W_CAIN);
    bf16_t* wCaOut = (bf16_t*)(ws + OFF_W_CAOUT);
    bf16_t* wFc1   = (bf16_t*)(ws + OFF_W_FC1);
    bf16_t* wGate  = (bf16_t*)(ws + OFF_W_GATE);
    bf16_t* wFc2   = (bf16_t*)(ws + OFF_W_FC2);
    float*  X1     = (float*)(ws + OFF_X1);
    float*  X2     = (float*)(ws + OFF_X2);
    bf16_t* XN     = (bf16_t*)(ws + OFF_XN);
    float*  QKV    = (float*)(ws + OFF_QKV);
    bf16_t* QH     = (bf16_t*)(ws + OFF_QH);
    bf16_t* KH     = (bf16_t*)(ws + OFF_KH);
    bf16_t* VT     = (bf16_t*)(ws + OFF_VT);
    bf16_t* TBUF   = (bf16_t*)(ws + OFF_TBUF);
    bf16_t* OBUF   = (bf16_t*)(ws + OFF_OBUF);
    bf16_t* QCB    = (bf16_t*)(ws + OFF_QCB);
    bf16_t* KVB    = (bf16_t*)(ws + OFF_KVB);
    bf16_t* VTC    = (bf16_t*)(ws + OFF_VTC);
    float*  HBUF   = (float*)(ws + OFF_HBUF);
    bf16_t* HGB    = (bf16_t*)(ws + OFF_HGB);

    const int M = 4096;                                   // B*S
    const float kscale = 0.125f * 1.44269504088896f;      // 1/sqrt(64) * log2(e)
    dim3 blk(256);

    // ---- weights -> bf16 (deterministic, every call) ----
    auto cvt = [&](const float* src, bf16_t* dst, int n) {
        cvt_bf16_kernel<<<dim3((n + 255) / 256), blk, 0, stream>>>(src, dst, n);
    };
    cvt(attn_in_w, wAttn, 3 * 1024 * 1024);
    cvt(ca_in_w,   wCaIn, 3 * 1024 * 1024);
    cvt(ca_out_w,  wCaOut, 1024 * 1024);
    cvt(fc1_w,  wFc1,  4 * 1024 * 1024);
    cvt(gate_w, wGate, 4 * 1024 * 1024);
    cvt(fc2_w,  wFc2,  4 * 1024 * 1024);

    // ---- phase A: self-attention ----
    ln_bf16_kernel<<<dim3(M), blk, 0, stream>>>(x, ln1_g, ln1_b, XN);
    gemm_bf16_kernel<0><<<dim3(3072 / 128, M / 128), blk, 0, stream>>>(
        XN, wAttn, attn_in_b, nullptr, QKV, nullptr, M, 3072, 1024);
    rope_prep_kernel<<<dim3((4 * 1024 * 1024) / 256), blk, 0, stream>>>(QKV, rot, QH, KH, VT);
    attn_kernel<0><<<dim3(1024 / 128, 64), blk, 0, stream>>>(
        QH, KH, VT, x, X1, nullptr, 1024, 1024, 1024, 1024L * 1024L, kscale);

    // ---- phase B: cross-attention ----
    ln_bf16_kernel<<<dim3(M), blk, 0, stream>>>(X1, ln2_g, ln2_b, XN);
    gemm_bf16_kernel<1><<<dim3(1024 / 128, M / 128), blk, 0, stream>>>(
        XN, wCaIn, ca_in_b, nullptr, nullptr, QCB, M, 1024, 1024);
    cvt(text_emb, TBUF, 1024 * 1024);
    gemm_bf16_kernel<1><<<dim3(2048 / 128, 1024 / 128), blk, 0, stream>>>(
        TBUF, wCaIn + (size_t)1024 * 1024, ca_in_b + 1024, nullptr, nullptr, KVB,
        1024, 2048, 1024);
    cross_vt_kernel<<<dim3((1024 * 1024) / 256), blk, 0, stream>>>(KVB, VTC);
    attn_kernel<1><<<dim3(1024 / 128, 64), blk, 0, stream>>>(
        QCB, KVB, VTC, nullptr, nullptr, OBUF, 1024, 256, 2048, 256L * 2048L, kscale);
    gemm_bf16_kernel<4><<<dim3(1024 / 128, M / 128), blk, 0, stream>>>(
        OBUF, wCaOut, ca_out_b, X1, X2, nullptr, M, 1024, 1024);

    // ---- phase C: gated MLP ----
    ln_bf16_kernel<<<dim3(M), blk, 0, stream>>>(X2, ln3_g, ln3_b, XN);
    gemm_bf16_kernel<2><<<dim3(4096 / 128, M / 128), blk, 0, stream>>>(
        XN, wFc1, fc1_b, nullptr, HBUF, nullptr, M, 4096, 1024);
    gemm_bf16_kernel<3><<<dim3(4096 / 128, M / 128), blk, 0, stream>>>(
        XN, wGate, gate_b, HBUF, nullptr, HGB, M, 4096, 1024);
    gemm_bf16_kernel<4><<<dim3(1024 / 128, M / 128), blk, 0, stream>>>(
        HGB, wFc2, fc2_b, X2, out, nullptr, M, 1024, 4096);
}
